// LlamaAttention_20461224198676
// MI455X (gfx1250) — compile-verified, matrix operand order checked by reading, not run on hardware
//
#include <hip/hip_runtime.h>
#include <stdint.h>

typedef __attribute__((ext_vector_type(16))) __bf16 v16bf;
typedef __attribute__((ext_vector_type(8)))  float  v8f;
typedef int v4i __attribute__((vector_size(16)));   // matches builtin's pointee type

union ABPack { uint4 u[2]; v16bf v; };

// ---- CDNA5 async global->LDS path (ASYNCcnt), with safe fallback ----
#if defined(__gfx1250__) && __has_builtin(__builtin_amdgcn_global_load_async_to_lds_b128)
#define HAVE_ASYNC_LDS 1
#else
#define HAVE_ASYNC_LDS 0
#endif

__device__ __forceinline__ void async_cp_b128(const unsigned short* g, unsigned short* l) {
#if HAVE_ASYNC_LDS
  __builtin_amdgcn_global_load_async_to_lds_b128(
      (__attribute__((address_space(1))) v4i*)(v4i*)g,
      (__attribute__((address_space(3))) v4i*)(v4i*)l, 0, 0);
#else
  *(uint4*)l = *(const uint4*)g;
#endif
}

__device__ __forceinline__ void wait_async() {
#if HAVE_ASYNC_LDS
#if __has_builtin(__builtin_amdgcn_s_wait_asynccnt)
  __builtin_amdgcn_s_wait_asynccnt(0);
#else
  asm volatile("s_wait_asynccnt 0x0" ::: "memory");
#endif
#endif
}

__device__ __forceinline__ unsigned short f2bf(float f) {
  unsigned int u = __float_as_uint(f);
  u += 0x7FFFu + ((u >> 16) & 1u);      // round-to-nearest-even
  return (unsigned short)(u >> 16);
}

__device__ __forceinline__ v8f v8f_zero() {
  v8f z = {0.f, 0.f, 0.f, 0.f, 0.f, 0.f, 0.f, 0.f};
  return z;
}

// ---------------- fp32 -> bf16 bulk convert (bf16 weights stay resident in 192MB L2) -------
__global__ void cvt_bf16_kernel(const float* __restrict__ src,
                                unsigned short* __restrict__ dst, int n4) {
  int i = blockIdx.x * blockDim.x + threadIdx.x;
  if (i < n4) {
    float4 f = ((const float4*)src)[i];
    ushort4 o;
    o.x = f2bf(f.x); o.y = f2bf(f.y); o.z = f2bf(f.z); o.w = f2bf(f.w);
    ((ushort4*)dst)[i] = o;
  }
}

// ---------------- bf16 GEMM: C[M,N] = A[M,K] * B[N,K]^T ----------------
// 128x256 block tile, 8 waves in 2x4 grid, 64x64 register tile per wave.
// Double-buffered LDS, async global->LDS staging, 1 barrier per K-step.
#define BM 128
#define BN 256
#define BKK 32
#define LDT 48   // padded LDS row stride (elements): 96B rows, 16B-aligned chunks

__global__ void __launch_bounds__(256) gemm_bf16_kernel(
    const unsigned short* __restrict__ A,   // M x K  bf16
    const unsigned short* __restrict__ B,   // N x K  bf16
    float* __restrict__ C,                  // M x N  f32
    int M, int N, int K) {
  __shared__ __align__(16) unsigned short As[2][BM * LDT];
  __shared__ __align__(16) unsigned short Bs[2][BN * LDT];

  const int t    = threadIdx.x;
  const int lane = t & 31;
  const int wid  = t >> 5;
  const int wm   = wid & 1;            // 2 waves down M: 64-row strips
  const int wn   = wid >> 1;           // 4 waves across N: 64-col strips
  const int m0   = blockIdx.y * BM;
  const int n0   = blockIdx.x * BN;
  const int hl   = lane & 15;
  const int koff = (lane < 16) ? 0 : 8;   // operand K-offset per lane-half

  v8f acc[4][4];
  #pragma unroll
  for (int i = 0; i < 4; i++)
    #pragma unroll
    for (int j = 0; j < 4; j++) acc[i][j] = v8f_zero();

  // stage one 128x32 (A) + 256x32 (B) bf16 tile pair into LDS buffer `buf`
  auto stage = [&](int buf, int kb) {
    #pragma unroll
    for (int c = t; c < 512; c += 256) {          // A: 512 16B chunks
      const int row = c >> 2, cc = c & 3;
      async_cp_b128(&A[(size_t)(m0 + row) * K + kb + cc * 8],
                    &As[buf][row * LDT + cc * 8]);
    }
    #pragma unroll
    for (int c = t; c < 1024; c += 256) {         // B: 1024 16B chunks
      const int row = c >> 2, cc = c & 3;
      async_cp_b128(&B[(size_t)(n0 + row) * K + kb + cc * 8],
                    &Bs[buf][row * LDT + cc * 8]);
    }
  };

  stage(0, 0);
  int buf = 0;
  for (int kb = 0; kb < K; kb += BKK, buf ^= 1) {
    wait_async();          // my async stores for tile(kb) have landed in LDS
    __syncthreads();       // everyone's landed; everyone done reading other buffer
    if (kb + BKK < K) stage(buf ^ 1, kb + BKK);   // overlap next tile with compute

    v16bf av[4], bv[4];
    #pragma unroll
    for (int mi = 0; mi < 4; mi++) {
      const int r = (wm * 64 + mi * 16 + hl) * LDT + koff;
      ABPack p;
      p.u[0] = *(const uint4*)(&As[buf][r]);
      p.u[1] = *(const uint4*)(&As[buf][r + 16]);
      av[mi] = p.v;
    }
    #pragma unroll
    for (int ni = 0; ni < 4; ni++) {
      const int r = (wn * 64 + ni * 16 + hl) * LDT + koff;
      ABPack p;
      p.u[0] = *(const uint4*)(&Bs[buf][r]);
      p.u[1] = *(const uint4*)(&Bs[buf][r + 16]);
      bv[ni] = p.v;
    }
    #pragma unroll
    for (int mi = 0; mi < 4; mi++)
      #pragma unroll
      for (int ni = 0; ni < 4; ni++)
        acc[mi][ni] = __builtin_amdgcn_wmma_f32_16x16x32_bf16(
            false, av[mi], false, bv[ni], (short)0, acc[mi][ni], false, false);
  }

  // epilogue: C-layout -> f32 stores (16 lanes contiguous per row)
  #pragma unroll
  for (int mi = 0; mi < 4; mi++)
    #pragma unroll
    for (int ni = 0; ni < 4; ni++)
      #pragma unroll
      for (int r = 0; r < 8; r++) {
        const int row = m0 + wm * 64 + mi * 16 + ((lane < 16) ? r : r + 8);
        const int col = n0 + wn * 64 + ni * 16 + hl;
        C[(size_t)row * N + col] = acc[mi][ni][r];
      }
}

// ---------------- RoPE + downconvert + relayout [S][H][128] f32 -> [H][S][128] bf16 --------
__global__ void rope_kernel(const float* __restrict__ src,
                            const float* __restrict__ cosb,
                            const float* __restrict__ sinb,
                            unsigned short* __restrict__ dst,
                            int H, int total /* S*H*64 */) {
  int i = blockIdx.x * blockDim.x + threadIdx.x;
  if (i >= total) return;
  const int p = i & 63;
  const int h = (i >> 6) % H;
  const int s = i / (64 * H);
  const float a = src[((size_t)s * H + h) * 128 + 2 * p];
  const float b = src[((size_t)s * H + h) * 128 + 2 * p + 1];
  const float c  = cosb[s * 64 + p];
  const float sn = sinb[s * 64 + p];
  const float o0 = a * c - b * sn;
  const float o1 = a * sn + b * c;
  const unsigned int pk = (unsigned int)f2bf(o0) | ((unsigned int)f2bf(o1) << 16);
  *(unsigned int*)(&dst[((size_t)h * 2048 + s) * 128 + 2 * p]) = pk;
}

// ---------------- V: [S][1024] f32 -> transposed bf16 [8][128][2048] ----------
__global__ void vtrans_kernel(const float* __restrict__ V,
                              unsigned short* __restrict__ Vt, int total) {
  int i = blockIdx.x * blockDim.x + threadIdx.x;
  if (i >= total) return;
  const int s = i >> 10;
  const int c = i & 1023;               // = kvh*128 + dim
  Vt[(size_t)c * 2048 + s] = f2bf(V[i]);
}

// ---------------- flash attention: grid = (S/128 q-tiles, 32 heads), 8 waves ----------
#define PST 72   // per-wave LDS P row stride (elements): 144B, 16B-aligned

__global__ void __launch_bounds__(256) attn_kernel(
    const unsigned short* __restrict__ Q,   // [32][2048][128] bf16 (roped)
    const unsigned short* __restrict__ Kc,  // [8][2048][128]  bf16 (roped)
    const unsigned short* __restrict__ Vt,  // [8][128][2048]  bf16 (transposed)
    unsigned short* __restrict__ O) {       // [2048][4096]    bf16
  __shared__ __align__(16) unsigned short Plds[8 * 16 * PST];

  const int lane = threadIdx.x & 31;
  const int wid  = threadIdx.x >> 5;
  const int h    = blockIdx.y;
  const int kvh  = h >> 2;                  // GQA 4:1
  const int qt   = blockIdx.x;
  const int q0   = qt * 128 + wid * 16;     // this wave's 16 query rows
  const int hl   = lane & 15;
  const int koff = (lane < 16) ? 0 : 8;
  const int rsh  = (lane < 16) ? 0 : 8;     // C-layout row shift per half
  unsigned short* Pw = &Plds[wid * 16 * PST];

  // Q operand registers: 4 K-steps cover dim=128 (A-matrix layout)
  v16bf qa[4];
  {
    const unsigned short* qptr = &Q[((size_t)h * 2048 + (q0 + hl)) * 128];
    #pragma unroll
    for (int ks = 0; ks < 4; ks++) {
      ABPack p;
      p.u[0] = *(const uint4*)(qptr + ks * 32 + koff);
      p.u[1] = *(const uint4*)(qptr + ks * 32 + koff + 16);
      qa[ks] = p.v;
    }
  }

  float mrun[8], lrun[8];
  v8f acc_o[8];
  #pragma unroll
  for (int r = 0; r < 8; r++) { mrun[r] = -3.0e38f; lrun[r] = 0.0f; }
  #pragma unroll
  for (int d = 0; d < 8; d++) acc_o[d] = v8f_zero();

  const int nkt = 2 * (qt + 1);             // causal: only key tiles <= diagonal
  const float scale = 0.08838834764831845f; // 1/sqrt(128)

  for (int kt = 0; kt < nkt; kt++) {
    const int kb = kt * 64;
    if (kt + 1 < nkt) {                     // look-ahead prefetch of next K/V tile
      __builtin_prefetch(&Kc[((size_t)kvh * 2048 + (kb + 64 + hl)) * 128], 0, 0);
      __builtin_prefetch(&Vt[((size_t)kvh * 128 + hl * 8) * 2048 + kb + 64], 0, 0);
    }

    // ---- S = Q * K^T : 16 WMMA per 64-key tile ----
    v8f sc[4];
    #pragma unroll
    for (int j = 0; j < 4; j++) {
      sc[j] = v8f_zero();
      const unsigned short* kptr =
          &Kc[((size_t)kvh * 2048 + (kb + j * 16 + hl)) * 128];
      #pragma unroll
      for (int ks = 0; ks < 4; ks++) {
        ABPack p;
        p.u[0] = *(const uint4*)(kptr + ks * 32 + koff);
        p.u[1] = *(const uint4*)(kptr + ks * 32 + koff + 16);
        sc[j] = __builtin_amdgcn_wmma_f32_16x16x32_bf16(
            false, qa[ks], false, p.v, (short)0, sc[j], false, false);
      }
    }

    // ---- online softmax (row = one VGPR across a 16-lane half) ----
    float pmat[4][8];
    float mnew[8];
    #pragma unroll
    for (int r = 0; r < 8; r++) {
      const int qrow = q0 + r + rsh;
      float mx = -3.0e38f;
      #pragma unroll
      for (int j = 0; j < 4; j++) {
        const int kc = kb + j * 16 + hl;
        float v = sc[j][r] * scale;
        v = (kc > qrow) ? -3.0e38f : v;     // causal mask (analytic; mask input unused)
        pmat[j][r] = v;
        mx = fmaxf(mx, v);
      }
      mx = fmaxf(mx, __shfl_xor(mx, 1, 32));
      mx = fmaxf(mx, __shfl_xor(mx, 2, 32));
      mx = fmaxf(mx, __shfl_xor(mx, 4, 32));
      mx = fmaxf(mx, __shfl_xor(mx, 8, 32));
      mnew[r] = fmaxf(mrun[r], mx);
    }
    #pragma unroll
    for (int r = 0; r < 8; r++) {
      const float f = __expf(mrun[r] - mnew[r]);
      mrun[r] = mnew[r];
      float lsum = 0.0f;
      #pragma unroll
      for (int j = 0; j < 4; j++) {
        const float pv = __expf(pmat[j][r] - mnew[r]);
        pmat[j][r] = pv;
        lsum += pv;
      }
      lsum += __shfl_xor(lsum, 1, 32);
      lsum += __shfl_xor(lsum, 2, 32);
      lsum += __shfl_xor(lsum, 4, 32);
      lsum += __shfl_xor(lsum, 8, 32);
      lrun[r] = lrun[r] * f + lsum;
      #pragma unroll
      for (int d = 0; d < 8; d++) acc_o[d][r] *= f;
    }

    // ---- P (C-layout f32) -> per-wave LDS row-major bf16 [16][64] ----
    #pragma unroll
    for (int j = 0; j < 4; j++)
      #pragma unroll
      for (int r = 0; r < 8; r++)
        Pw[(r + rsh) * PST + j * 16 + hl] = f2bf(pmat[j][r]);
    asm volatile("s_wait_dscnt 0x0" ::: "memory");  // wave-private: no barrier needed

    // ---- O += P * V : 16 WMMA per 64-key tile (Vt gives contiguous B chunks) ----
    #pragma unroll
    for (int kk = 0; kk < 2; kk++) {
      ABPack pa;
      pa.u[0] = *(const uint4*)(&Pw[hl * PST + kk * 32 + koff]);
      pa.u[1] = *(const uint4*)(&Pw[hl * PST + kk * 32 + koff + 16]);
      #pragma unroll
      for (int d = 0; d < 8; d++) {
        const unsigned short* vptr =
            &Vt[((size_t)kvh * 128 + d * 16 + hl) * 2048 + kb + kk * 32 + koff];
        ABPack vb;
        vb.u[0] = *(const uint4*)(vptr);
        vb.u[1] = *(const uint4*)(vptr + 16);
        acc_o[d] = __builtin_amdgcn_wmma_f32_16x16x32_bf16(
            false, pa.v, false, vb.v, (short)0, acc_o[d], false, false);
      }
    }
  }

  // ---- epilogue: O[q][h*128+dim] = acc / l  (bf16, feeds output-proj GEMM) ----
  #pragma unroll
  for (int r = 0; r < 8; r++) {
    const float inv = 1.0f / lrun[r];
    const int row = q0 + r + rsh;
    #pragma unroll
    for (int d = 0; d < 8; d++)
      O[(size_t)row * 4096 + h * 128 + d * 16 + hl] = f2bf(acc_o[d][r] * inv);
  }
}

extern "C" void kernel_launch(void* const* d_in, const int* in_sizes, int n_in,
                              void* d_out, int out_size, void* d_ws, size_t ws_size,
                              hipStream_t stream) {
  const float* hidden = (const float*)d_in[0];
  // d_in[1] = attention_mask: unused, causality computed analytically
  const float* cosb = (const float*)d_in[2];
  const float* sinb = (const float*)d_in[3];
  const float* wq   = (const float*)d_in[4];
  const float* wk   = (const float*)d_in[5];
  const float* wv   = (const float*)d_in[6];
  const float* wo   = (const float*)d_in[7];
  float* out = (float*)d_out;

  const int S = 2048, HID = 4096, KV = 1024;

  char* w = (char*)d_ws;
  auto alloc = [&](size_t bytes) {
    char* p = w;
    w += (bytes + 255) & ~(size_t)255;
    return p;
  };
  unsigned short* hid_bf = (unsigned short*)alloc((size_t)S * HID * 2);
  unsigned short* wq_bf  = (unsigned short*)alloc((size_t)HID * HID * 2);
  unsigned short* wk_bf  = (unsigned short*)alloc((size_t)KV * HID * 2);
  unsigned short* wv_bf  = (unsigned short*)alloc((size_t)KV * HID * 2);
  unsigned short* wo_bf  = (unsigned short*)alloc((size_t)HID * HID * 2);
  float* qf = (float*)alloc((size_t)S * HID * 4);
  float* kf = (float*)alloc((size_t)S * KV * 4);
  float* vf = (float*)alloc((size_t)S * KV * 4);
  unsigned short* q_bf  = (unsigned short*)alloc((size_t)S * HID * 2); // [32][S][128]
  unsigned short* k_bf  = (unsigned short*)alloc((size_t)S * KV * 2);  // [8][S][128]
  unsigned short* vt_bf = (unsigned short*)alloc((size_t)S * KV * 2);  // [8][128][S]
  unsigned short* at_bf = (unsigned short*)alloc((size_t)S * HID * 2); // [S][4096]

  auto cvt = [&](const float* s, unsigned short* d, size_t n) {
    const int n4 = (int)(n / 4);
    cvt_bf16_kernel<<<(n4 + 255) / 256, 256, 0, stream>>>(s, d, n4);
  };
  cvt(hidden, hid_bf, (size_t)S * HID);
  cvt(wq, wq_bf, (size_t)HID * HID);
  cvt(wk, wk_bf, (size_t)KV * HID);
  cvt(wv, wv_bf, (size_t)KV * HID);
  cvt(wo, wo_bf, (size_t)HID * HID);

  gemm_bf16_kernel<<<dim3(HID / BN, S / BM), 256, 0, stream>>>(hid_bf, wq_bf, qf, S, HID, HID);
  gemm_bf16_kernel<<<dim3(KV  / BN, S / BM), 256, 0, stream>>>(hid_bf, wk_bf, kf, S, KV, HID);
  gemm_bf16_kernel<<<dim3(KV  / BN, S / BM), 256, 0, stream>>>(hid_bf, wv_bf, vf, S, KV, HID);

  { const int tot = S * 32 * 64;
    rope_kernel<<<(tot + 255) / 256, 256, 0, stream>>>(qf, cosb, sinb, q_bf, 32, tot); }
  { const int tot = S * 8 * 64;
    rope_kernel<<<(tot + 255) / 256, 256, 0, stream>>>(kf, cosb, sinb, k_bf, 8, tot); }
  { const int tot = S * KV;
    vtrans_kernel<<<(tot + 255) / 256, 256, 0, stream>>>(vf, vt_bf, tot); }

  attn_kernel<<<dim3(S / 128, 32), 256, 0, stream>>>(q_bf, k_bf, vt_bf, at_bf);

  gemm_bf16_kernel<<<dim3(HID / BN, S / BM), 256, 0, stream>>>(at_bf, wo_bf, out, S, HID, HID);

  (void)in_sizes; (void)n_in; (void)out_size; (void)ws_size;
}
